// QuantumUncertaintyHead_68418829025797
// MI455X (gfx1250) — matrix operand order repaired and checked by reading, not compiled, not run on hardware
//
#include <hip/hip_runtime.h>
#include <math.h>

// ---------------------------------------------------------------------------
// QuantumUncertaintyHead for MI455X (gfx1250, wave32)
//   K1: x[1024,768] @ Wr^T[768,10] via v_wmma_f32_16x16x32_f16, + br, tanh, *pi/2
//       (unconditional float4 loads + multiplicative zero-mask: no divergence)
//   K2: 10-qubit statevector sim, one wave per batch element, state in VGPRs
//       (32 complex amps/lane = 64 VGPRs), cross-lane gates via __shfl_xor.
// ---------------------------------------------------------------------------

typedef __attribute__((ext_vector_type(16))) _Float16 v16h;
typedef __attribute__((ext_vector_type(8)))  float    v8f;

#define N_QUBITS   10
#define N_LAYERS   4
#define INPUT_DIM  768
#define BATCH      1024
#define RED_STRIDE 16   // padded column stride for reduced angles in d_ws

// ------------------------- Kernel 1: GEMM + tanh ---------------------------
// One wave per 16-batch tile. A = x tile (16x32 f16), B = Wr^T tile (32x16 f16,
// cols >= 10 zero-masked), C accumulates f32 over 24 K-chunks.
__global__ __launch_bounds__(32) void qhead_gemm_tanh(
    const float* __restrict__ x, const float* __restrict__ Wr,
    const float* __restrict__ br, float* __restrict__ reduced) {
  const int lane = threadIdx.x;        // 0..31
  const int l15  = lane & 15;
  const int hsel = lane >> 4;          // lane half select
  const int b0   = blockIdx.x * 16;    // batch tile base

  // A-matrix (16x32 f16): lane half 0 -> K 0..7 / 16..23 ; half 1 -> 8..15 / 24..31
  const int koffA = hsel * 8;
  // B-matrix (32x16 f16): lane half 0 -> K 0..7 / 16..23 ; half 1 -> 8..15 / 24..31
  const int koffB = hsel * 8;

  const float* xrow  = x + (size_t)(b0 + l15) * INPUT_DIM;
  const int    qrow  = (l15 < N_QUBITS) ? l15 : 0;      // clamp to valid row
  const float  wmask = (l15 < N_QUBITS) ? 1.0f : 0.0f;  // zero-pad cols 10..15
  const float* wrow  = Wr + (size_t)qrow * INPUT_DIM;

  v8f acc = {};
  for (int kc = 0; kc < INPUT_DIM; kc += 32) {
    // All offsets are 16B-aligned (row bases are 768-float aligned,
    // kc % 32 == 0, koffA in {0,8}, koffB in {0,8}).
    const float4 a0 = *(const float4*)(xrow + kc + koffA);
    const float4 a1 = *(const float4*)(xrow + kc + koffA + 4);
    const float4 a2 = *(const float4*)(xrow + kc + koffA + 16);
    const float4 a3 = *(const float4*)(xrow + kc + koffA + 20);
    const float4 w0 = *(const float4*)(wrow + kc + koffB);
    const float4 w1 = *(const float4*)(wrow + kc + koffB + 4);
    const float4 w2 = *(const float4*)(wrow + kc + koffB + 16);
    const float4 w3 = *(const float4*)(wrow + kc + koffB + 20);

    v16h a, bm;
    a[0]  = (_Float16)a0.x; a[1]  = (_Float16)a0.y;
    a[2]  = (_Float16)a0.z; a[3]  = (_Float16)a0.w;
    a[4]  = (_Float16)a1.x; a[5]  = (_Float16)a1.y;
    a[6]  = (_Float16)a1.z; a[7]  = (_Float16)a1.w;
    a[8]  = (_Float16)a2.x; a[9]  = (_Float16)a2.y;
    a[10] = (_Float16)a2.z; a[11] = (_Float16)a2.w;
    a[12] = (_Float16)a3.x; a[13] = (_Float16)a3.y;
    a[14] = (_Float16)a3.z; a[15] = (_Float16)a3.w;

    bm[0]  = (_Float16)(w0.x * wmask); bm[1]  = (_Float16)(w0.y * wmask);
    bm[2]  = (_Float16)(w0.z * wmask); bm[3]  = (_Float16)(w0.w * wmask);
    bm[4]  = (_Float16)(w1.x * wmask); bm[5]  = (_Float16)(w1.y * wmask);
    bm[6]  = (_Float16)(w1.z * wmask); bm[7]  = (_Float16)(w1.w * wmask);
    bm[8]  = (_Float16)(w2.x * wmask); bm[9]  = (_Float16)(w2.y * wmask);
    bm[10] = (_Float16)(w2.z * wmask); bm[11] = (_Float16)(w2.w * wmask);
    bm[12] = (_Float16)(w3.x * wmask); bm[13] = (_Float16)(w3.y * wmask);
    bm[14] = (_Float16)(w3.z * wmask); bm[15] = (_Float16)(w3.w * wmask);

    // (neg_a, A, neg_b, B, c_mod, C, reuse_a, reuse_b)
    acc = __builtin_amdgcn_wmma_f32_16x16x32_f16(false, a, false, bm,
                                                 (short)0, acc, false, false);
  }

  // C/D layout: VGPR r -> row (b0 + hsel*8 + r), col = lane&15
  const float bias    = br[qrow] * wmask;
  const float HALF_PI = 1.57079632679489662f;
#pragma unroll
  for (int r = 0; r < 8; ++r) {
    int row = b0 + hsel * 8 + r;
    reduced[row * RED_STRIDE + l15] = tanhf(acc[r] + bias) * HALF_PI;
  }
}

// --------------------- Kernel 2: quantum circuit sim -----------------------
// Amplitude index i = (lane << 5) | r.  Wire w maps to index bit p = 9 - w
// (row-major flatten of [2]*10).  p in 0..4 -> register bit (compile-time
// after unroll); p in 5..9 -> lane bit (shuffle / per-lane select).
__global__ __launch_bounds__(256) void qhead_qsim(
    const float* __restrict__ reduced, const float* __restrict__ weights,
    const float* __restrict__ Wo, const float* __restrict__ bo,
    float* __restrict__ out) {
  const int lane = threadIdx.x & 31;
  const int b    = (blockIdx.x * 256 + threadIdx.x) >> 5;  // 0..1023

  float sr[32], si[32];
#pragma unroll
  for (int r = 0; r < 32; ++r) { sr[r] = 0.0f; si[r] = 0.0f; }
  if (lane == 0) sr[0] = 1.0f;  // |0...0>

  auto ry_gate = [&](int w, float th) {
    float c = __cosf(0.5f * th), s = __sinf(0.5f * th);
    int p = 9 - w;
    if (p >= 5) {                       // lane-bit qubit: butterfly across lanes
      int tm = 1 << (p - 5);
      float t = (lane & tm) ? s : -s;   // bit0: c*a0 - s*a1 ; bit1: s*a0 + c*a1
#pragma unroll
      for (int r = 0; r < 32; ++r) {
        float pr = __shfl_xor(sr[r], tm, 32);
        float pi = __shfl_xor(si[r], tm, 32);
        sr[r] = c * sr[r] + t * pr;
        si[r] = c * si[r] + t * pi;
      }
    } else {                            // register-bit qubit: in-lane pairs
      int m = 1 << p;
#pragma unroll
      for (int r = 0; r < 32; ++r) {
        if ((r & m) == 0) {
          int r1 = r | m;
          float a0r = sr[r], a0i = si[r], a1r = sr[r1], a1i = si[r1];
          sr[r]  = c * a0r - s * a1r;  si[r]  = c * a0i - s * a1i;
          sr[r1] = s * a0r + c * a1r;  si[r1] = s * a0i + c * a1i;
        }
      }
    }
  };

  auto rz_gate = [&](int w, float th) { // diagonal: never needs communication
    float c = __cosf(0.5f * th), s = __sinf(0.5f * th);
    int p = 9 - w;
    if (p >= 5) {
      int lm = 1 << (p - 5);
      float t = (lane & lm) ? -s : s;   // e^{-i th/2} on bit0, e^{+i th/2} on bit1
#pragma unroll
      for (int r = 0; r < 32; ++r) {
        float xr = sr[r], xi = si[r];
        sr[r] = c * xr + t * xi;
        si[r] = c * xi - t * xr;
      }
    } else {
      int m = 1 << p;
#pragma unroll
      for (int r = 0; r < 32; ++r) {
        float t = (r & m) ? -s : s;
        float xr = sr[r], xi = si[r];
        sr[r] = c * xr + t * xi;
        si[r] = c * xi - t * xr;
      }
    }
  };

  auto cnot_gate = [&](int cw, int tw) {
    int pc = 9 - cw, pt = 9 - tw;
    if (pc >= 5 && pt >= 5) {           // ctrl & tgt in lane bits
      int cm = 1 << (pc - 5), tm = 1 << (pt - 5);
      bool ctl = (lane & cm) != 0;
#pragma unroll
      for (int r = 0; r < 32; ++r) {
        float er = __shfl_xor(sr[r], tm, 32);
        float ei = __shfl_xor(si[r], tm, 32);
        sr[r] = ctl ? er : sr[r];
        si[r] = ctl ? ei : si[r];
      }
    } else if (pc >= 5) {               // ctrl lane bit, tgt register bit
      int cm = 1 << (pc - 5), m = 1 << pt;
      bool ctl = (lane & cm) != 0;
#pragma unroll
      for (int r = 0; r < 32; ++r) {
        if ((r & m) == 0) {
          int r1 = r | m;
          float ar = sr[r], brr = sr[r1], ai = si[r], bi = si[r1];
          sr[r]  = ctl ? brr : ar;  sr[r1] = ctl ? ar : brr;
          si[r]  = ctl ? bi  : ai;  si[r1] = ctl ? ai : bi;
        }
      }
    } else if (pt >= 5) {               // ctrl register bit (compile-time), tgt lane bit
      int m = 1 << pc, tm = 1 << (pt - 5);
#pragma unroll
      for (int r = 0; r < 32; ++r) {
        if (r & m) {                    // unconditional swap across lanes for these regs
          sr[r] = __shfl_xor(sr[r], tm, 32);
          si[r] = __shfl_xor(si[r], tm, 32);
        }
      }
    } else {                            // both register bits: pure register rename
      int cm = 1 << pc, tm = 1 << pt;
#pragma unroll
      for (int r = 0; r < 32; ++r) {
        if ((r & cm) && !(r & tm)) {
          int r1 = r | tm;
          float t0 = sr[r]; sr[r] = sr[r1]; sr[r1] = t0;
          float t1 = si[r]; si[r] = si[r1]; si[r1] = t1;
        }
      }
    }
  };

  // AngleEmbedding RY (per-batch angles, wave-uniform)
  float ang[N_QUBITS];
#pragma unroll
  for (int q = 0; q < N_QUBITS; ++q) ang[q] = reduced[b * RED_STRIDE + q];
#pragma unroll
  for (int w = 0; w < N_QUBITS; ++w) ry_gate(w, ang[w]);

  // Layers: RY,RZ per wire then CNOT ring
#pragma unroll
  for (int l = 0; l < N_LAYERS; ++l) {
#pragma unroll
    for (int w = 0; w < N_QUBITS; ++w) {
      ry_gate(w, weights[(l * N_QUBITS + w) * 2 + 0]);
      rz_gate(w, weights[(l * N_QUBITS + w) * 2 + 1]);
    }
#pragma unroll
    for (int w = 0; w < N_QUBITS; ++w) cnot_gate(w, (w + 1) % N_QUBITS);
  }

  // Z expectations: z[w] = P(bit_{9-w}=0) - P(bit_{9-w}=1)
  float z[N_QUBITS];
  float total = 0.0f;
#pragma unroll
  for (int q = 5; q < N_QUBITS; ++q) z[q] = 0.0f;
#pragma unroll
  for (int r = 0; r < 32; ++r) {
    float p = sr[r] * sr[r] + si[r] * si[r];
    total += p;
#pragma unroll
    for (int w = 5; w < N_QUBITS; ++w) {
      int m = 1 << (9 - w);
      z[w] += (r & m) ? -p : p;
    }
  }
#pragma unroll
  for (int w = 0; w < 5; ++w) {
    int lm = 1 << (4 - w);
    z[w] = (lane & lm) ? -total : total;
  }
  // xor-butterfly reduction across the wave (all lanes end with the sum)
#pragma unroll
  for (int q = 0; q < N_QUBITS; ++q) {
    float v = z[q];
#pragma unroll
    for (int m2 = 1; m2 < 32; m2 <<= 1) v += __shfl_xor(v, m2, 32);
    z[q] = v;
  }

  if (lane == 0) {
    float logit = bo[0];
#pragma unroll
    for (int q = 0; q < N_QUBITS; ++q) logit += Wo[q] * z[q];
    out[b] = 1.0f / (1.0f + __expf(-logit));
  }
}

// ------------------------------ launcher -----------------------------------
extern "C" void kernel_launch(void* const* d_in, const int* in_sizes, int n_in,
                              void* d_out, int out_size, void* d_ws, size_t ws_size,
                              hipStream_t stream) {
  (void)in_sizes; (void)n_in; (void)out_size; (void)ws_size;
  const float* x       = (const float*)d_in[0];  // [1024,768]
  const float* Wr      = (const float*)d_in[1];  // [10,768]
  const float* br      = (const float*)d_in[2];  // [10]
  const float* weights = (const float*)d_in[3];  // [4,10,2]
  const float* Wo      = (const float*)d_in[4];  // [1,10]
  const float* bo      = (const float*)d_in[5];  // [1]
  float* out     = (float*)d_out;                // [1024]
  float* reduced = (float*)d_ws;                 // [1024,16] f32 scratch (64 KB)

  qhead_gemm_tanh<<<BATCH / 16, 32, 0, stream>>>(x, Wr, br, reduced);
  qhead_qsim<<<BATCH / 8, 256, 0, stream>>>(reduced, weights, Wo, bo, out);
}
